// Architecture_7481833030185
// MI455X (gfx1250) — compile-verified, matrix operand order checked by reading, not run on hardware
//
#include <hip/hip_runtime.h>
#include <hip/hip_bf16.h>

#define B_   32
#define S_   512
#define D_   512
#define H_   8
#define DK_  64
#define DF_  2048
#define MTOT (B_ * S_)   // 16384 rows

#define TM_  256         // GEMM workgroup tile M
#define TN_  128         // GEMM workgroup tile N

typedef __bf16 bf16_t;
typedef __attribute__((ext_vector_type(16))) __bf16 v16bf;
typedef __attribute__((ext_vector_type(8)))  __bf16 v8bf;
typedef __attribute__((ext_vector_type(8)))  float  v8f;

// ---------------------------------------------------------------------------
// 16x32 bf16 A-style WMMA fragment load from a row-major tile (global or LDS).
// ISA layout (CDNA5 ISA, 16-bit A 16x32): lanes 0-15 = rows 0-15 holding
// K 0..7 then 16..23; lanes 16-31 = rows 0-15 holding K 8..15 then 24..31.
// B fragments of a 32x16 matrix use the identical pattern applied to the
// transposed (16 rows x 32 cols) view, so one loader serves both.
// ---------------------------------------------------------------------------
static __device__ __forceinline__ v16bf load_frag16(const bf16_t* base, int ld) {
  const int lane = threadIdx.x & 31;
  const int row  = lane & 15;
  const int kh   = lane >> 4;
  const bf16_t* p = base + (size_t)row * ld + kh * 8;
  v8bf lo = *reinterpret_cast<const v8bf*>(p);
  v8bf hi = *reinterpret_cast<const v8bf*>(p + 16);
  return __builtin_shufflevector(lo, hi, 0, 1, 2, 3, 4, 5, 6, 7,
                                         8, 9, 10, 11, 12, 13, 14, 15);
}

static __device__ __forceinline__ v8f wmma_bf16(v16bf a, v16bf b, v8f c) {
  // emits v_wmma_f32_16x16x32_bf16
  return __builtin_amdgcn_wmma_f32_16x16x32_bf16(false, a, false, b,
                                                 (short)0, c, false, false);
}

// Per-lane async copy of 16 bytes global -> LDS (ASYNCcnt-tracked).
static __device__ __forceinline__ void async_b128(unsigned lds_addr,
                                                  const void* gaddr) {
  asm volatile("global_load_async_to_lds_b128 %0, %1, off"
               :: "v"(lds_addr), "v"((unsigned long long)(uintptr_t)gaddr)
               : "memory");
}

// ---------------------------------------------------------------------------
// GEMM: out[M,N] = A[M,K](bf16) * Wt[N,K](bf16, pre-transposed) + bias
// 256 threads = 8 waves (4M x 2N); WG tile 256x128; wave tile 64x64
// (4x4 WMMA tiles -> 16 WMMA per 32-K step).
// B (weight) tile staged in LDS via double-buffered async copies.
// ---------------------------------------------------------------------------
__global__ __launch_bounds__(256)
void gemm_bf16_kernel(const bf16_t* __restrict__ A, const bf16_t* __restrict__ Wt,
                      const float* __restrict__ bias,
                      float* __restrict__ outF, bf16_t* __restrict__ outB,
                      int M, int N, int K, int relu)
{
  __shared__ bf16_t sB[2][TN_ * 32];       // 2 x 8 KB weight tiles
  const int tid  = threadIdx.x;
  const int wave = tid >> 5;
  const int lane = tid & 31;
  const int mw = wave & 3;                 // 4 waves along M
  const int nw = wave >> 2;                // 2 waves along N
  const int m0 = blockIdx.y * TM_ + mw * 64;
  const int n0 = blockIdx.x * TN_;         // LDS tile covers [n0, n0+128)
  const int nloc = nw * 64;                // wave's N offset inside LDS tile
  const int nsteps = K >> 5;

  // Stage a 128(N) x 32(K) bf16 weight tile: 512 16-byte segments,
  // 2 per thread -> 2 async instructions per wave per tile.
  auto issue_tile = [&](int step, int buf) {
#pragma unroll
    for (int s = 0; s < 2; ++s) {
      const int seg = tid + s * 256;
      const int row = seg >> 2;            // 0..127
      const int c16 = (seg & 3) * 8;       // bf16 column: 0,8,16,24
      const bf16_t* g = Wt + (size_t)(n0 + row) * K + step * 32 + c16;
      const unsigned l = (unsigned)(uintptr_t)(&sB[buf][row * 32 + c16]);
      async_b128(l, g);
    }
  };

  v8f acc[4][4] = {};
  issue_tile(0, 0);

  for (int i = 0; i < nsteps; ++i) {
    const int buf = i & 1;
    if (i + 1 < nsteps) {
      issue_tile(i + 1, buf ^ 1);
      asm volatile("s_wait_asynccnt 0x2" ::: "memory");  // tile i landed
    } else {
      asm volatile("s_wait_asynccnt 0x0" ::: "memory");
    }
    __syncthreads();                       // all waves' stages visible

    const int k0 = i * 32;
    v16bf bfr[4];
#pragma unroll
    for (int j = 0; j < 4; ++j)            // B fragments from LDS (ds_load_b128)
      bfr[j] = load_frag16(&sB[buf][(nloc + j * 16) * 32], 32);

    if (i + 2 < nsteps)                    // prefetch A two tiles ahead
      __builtin_prefetch(A + (size_t)m0 * K + k0 + 64, 0, 1);

#pragma unroll
    for (int mi = 0; mi < 4; ++mi) {
      v16bf a = load_frag16(A + (size_t)(m0 + mi * 16) * K + k0, K);
#pragma unroll
      for (int j = 0; j < 4; ++j)
        acc[mi][j] = wmma_bf16(a, bfr[j], acc[mi][j]);
    }
    __syncthreads();                       // buf consumed; safe to refill later
  }

  // C/D layout: lane l -> N = l&15 ; VGPR r -> M = r + 8*(l>>4)
  const int nn = lane & 15;
  const int mh = (lane >> 4) * 8;
#pragma unroll
  for (int mi = 0; mi < 4; ++mi)
#pragma unroll
    for (int j = 0; j < 4; ++j) {
      const int ncol = n0 + nloc + j * 16 + nn;
      const float bv = bias ? bias[ncol] : 0.0f;
#pragma unroll
      for (int r = 0; r < 8; ++r) {
        const int mrow = m0 + mi * 16 + mh + r;
        float v = acc[mi][j][r] + bv;
        if (relu) v = v > 0.0f ? v : 0.0f;
        const size_t off = (size_t)mrow * N + ncol;
        if (outF) outF[off] = v;
        if (outB) outB[off] = (bf16_t)v;
      }
    }
}

// W[K,N] f32 -> Wt[N,K] bf16 (transpose + downconvert)
__global__ void wcast_kernel(const float* __restrict__ W, bf16_t* __restrict__ Wt,
                             int K, int N) {
  size_t idx = (size_t)blockIdx.x * blockDim.x + threadIdx.x;
  if (idx >= (size_t)K * N) return;
  int k = (int)(idx / N), n = (int)(idx % N);
  Wt[(size_t)n * K + k] = (bf16_t)W[idx];
}

// f32 -> {f32 copy, bf16 mirror}
__global__ void castcopy_kernel(const float* __restrict__ src,
                                float* __restrict__ dstF,
                                bf16_t* __restrict__ dstB, size_t n) {
  size_t i = (size_t)blockIdx.x * blockDim.x + threadIdx.x;
  if (i >= n) return;
  float v = src[i];
  if (dstF) dstF[i] = v;
  dstB[i] = (bf16_t)v;
}

// V (b,s,h*64+dk) bf16 -> Vt (b,h,dk,s) bf16 so P@V B-fragments are contiguous
__global__ void vtrans_kernel(const bf16_t* __restrict__ v, bf16_t* __restrict__ vt) {
  size_t i = (size_t)blockIdx.x * blockDim.x + threadIdx.x;
  if (i >= (size_t)B_ * S_ * D_) return;
  int d = (int)(i % D_);
  int s = (int)((i / D_) % S_);
  int b = (int)(i / ((size_t)D_ * S_));
  int h = d / DK_, dk = d % DK_;
  vt[(((size_t)b * H_ + h) * DK_ + dk) * S_ + s] = v[i];
}

// fused residual + LayerNorm, writes f32 + bf16 mirror. One row per block.
__global__ __launch_bounds__(128)
void ln_residual_kernel(const float* __restrict__ add, const float* __restrict__ res,
                        const float* __restrict__ g, const float* __restrict__ be,
                        float* __restrict__ outF, bf16_t* __restrict__ outB)
{
  const int row = blockIdx.x;
  const int tid = threadIdx.x;
  __shared__ float red[128];
  float v[4];
  float s = 0.f, s2 = 0.f;
#pragma unroll
  for (int j = 0; j < 4; ++j) {
    int d = tid + j * 128;
    float t = add[(size_t)row * D_ + d] + res[(size_t)row * D_ + d];
    v[j] = t; s += t; s2 += t * t;
  }
  red[tid] = s; __syncthreads();
  for (int off = 64; off > 0; off >>= 1) {
    if (tid < off) red[tid] += red[tid + off];
    __syncthreads();
  }
  const float mean = red[0] / (float)D_;
  __syncthreads();
  red[tid] = s2; __syncthreads();
  for (int off = 64; off > 0; off >>= 1) {
    if (tid < off) red[tid] += red[tid + off];
    __syncthreads();
  }
  const float var = red[0] / (float)D_ - mean * mean;
  const float rstd = rsqrtf(var + 1e-5f);
#pragma unroll
  for (int j = 0; j < 4; ++j) {
    int d = tid + j * 128;
    float o = (v[j] - mean) * rstd * g[d] + be[d];
    outF[(size_t)row * D_ + d] = o;
    outB[(size_t)row * D_ + d] = (bf16_t)o;
  }
}

// ---------------------------------------------------------------------------
// Forget-attention. One WG (4 waves / 128 thr) per (b, h, 16-query tile).
// Whole 16x512 score strip lives in LDS (32 KB). disttotal is analytically
// 1 (row has >=1 valid key) or 0 (fully masked), since softmax*mask sums to 1.
// ---------------------------------------------------------------------------
__global__ __launch_bounds__(128)
void attn_kernel(const bf16_t* __restrict__ qp, const bf16_t* __restrict__ kp,
                 const bf16_t* __restrict__ vt,
                 const float* __restrict__ ssim, const float* __restrict__ dec,
                 bf16_t* __restrict__ outB, int strict, int zero_pad)
{
  __shared__ float sS[16][S_];
  const int wg = blockIdx.x;
  const int qt = wg & 31;          // S_/16 tiles
  const int bh = wg >> 5;
  const int b  = bh / H_;
  const int h  = bh % H_;
  const int wave = threadIdx.x >> 5;
  const int lane = threadIdx.x & 31;
  const float NEGV = -3.0e38f;

  // ---- Phase 1: scores = Q K^T / sqrt(DK) -> LDS -------------------------
  const bf16_t* Qb = qp + ((size_t)(b * S_ + qt * 16) * D_) + h * DK_;
  v16bf aq0 = load_frag16(Qb + 0,  D_);
  v16bf aq1 = load_frag16(Qb + 32, D_);
  for (int kt = wave; kt < 32; kt += 4) {
    const bf16_t* Kb = kp + ((size_t)(b * S_ + kt * 16) * D_) + h * DK_;
    v8f acc = {};
    acc = wmma_bf16(aq0, load_frag16(Kb + 0,  D_), acc);
    acc = wmma_bf16(aq1, load_frag16(Kb + 32, D_), acc);
    const int n  = lane & 15;
    const int mh = (lane >> 4) * 8;
#pragma unroll
    for (int r = 0; r < 8; ++r)
      sS[mh + r][kt * 16 + n] = acc[r] * 0.125f;   // 1/sqrt(64)
  }
  __syncthreads();

  // ---- Phase 2: softmax -> scan -> decay -> softmax (per row, per wave) --
  for (int rr = 0; rr < 4; ++rr) {
    const int row = wave * 4 + rr;
    const int sg  = qt * 16 + row;
    float mx = NEGV;
    for (int k = lane; k < S_; k += 32) {
      bool msk = strict ? (k < sg) : (k <= sg);
      if (msk) mx = fmaxf(mx, sS[row][k]);
    }
    for (int off = 16; off > 0; off >>= 1) mx = fmaxf(mx, __shfl_xor(mx, off, 32));
    float sum = 0.f;
    for (int k = lane; k < S_; k += 32) {
      bool msk = strict ? (k < sg) : (k <= sg);
      sum += msk ? __expf(sS[row][k] - mx) : 0.f;
    }
    for (int off = 16; off > 0; off >>= 1) sum += __shfl_xor(sum, off, 32);
    const float inv       = sum > 0.f ? 1.f / sum : 0.f;
    const float disttotal = sum > 0.f ? 1.f : 0.f;

    float running = 0.f;
    for (int c0 = 0; c0 < S_; c0 += 32) {
      const int k = c0 + lane;
      bool msk = strict ? (k < sg) : (k <= sg);
      float raw = sS[row][k];
      float e = msk ? __expf(raw - mx) * inv : 0.f;
      float sc = e;                                   // wave32 inclusive scan
      for (int off = 1; off < 32; off <<= 1) {
        float t = __shfl_up(sc, off, 32);
        if (lane >= off) sc += t;
      }
      float distcum = running + sc;
      running += __shfl(sc, 31, 32);
      float pe = ssim[((size_t)b * S_ + sg) * S_ + k];
      float dd = (disttotal - distcum) * pe;
      float dist = dd > 0.f ? sqrtf(dd) : 0.f;
      float dv = dec[((size_t)b * S_ + sg) * S_ + k];
      float sp = dv > 20.f ? dv : log1pf(__expf(dv)); // softplus
      float te = __expf(-sp * dist);
      te = fminf(fmaxf(te, 1e-5f), 1e5f);
      sS[row][k] = msk ? raw * te : NEGV;
    }
    float mx2 = NEGV;
    for (int k = lane; k < S_; k += 32) mx2 = fmaxf(mx2, sS[row][k]);
    for (int off = 16; off > 0; off >>= 1) mx2 = fmaxf(mx2, __shfl_xor(mx2, off, 32));
    float sum2 = 0.f;
    for (int k = lane; k < S_; k += 32) sum2 += __expf(sS[row][k] - mx2);
    for (int off = 16; off > 0; off >>= 1) sum2 += __shfl_xor(sum2, off, 32);
    const float inv2 = 1.f / sum2;
    for (int k = lane; k < S_; k += 32) {
      float p = __expf(sS[row][k] - mx2) * inv2;
      if (zero_pad && sg == 0) p = 0.f;
      sS[row][k] = p;
    }
  }
  __syncthreads();

  // ---- Phase 3: O = P @ V (each wave owns a 16-wide dk slice) ------------
  const int n0 = wave * 16;
  const bf16_t* Vb = vt + ((size_t)bh * DK_ + n0) * S_;
  v8f acc = {};
  for (int k0 = 0; k0 < S_; k0 += 32) {
    const int row = lane & 15;
    const int cb  = k0 + (lane >> 4) * 8;
    v16bf a;
#pragma unroll
    for (int e = 0; e < 8; ++e) {
      a[e]     = (bf16_t)sS[row][cb + e];
      a[e + 8] = (bf16_t)sS[row][cb + 16 + e];
    }
    acc = wmma_bf16(a, load_frag16(Vb + k0, S_), acc);
  }
  const int n   = lane & 15;
  const int mh2 = (lane >> 4) * 8;
#pragma unroll
  for (int r = 0; r < 8; ++r) {
    const int srow = qt * 16 + mh2 + r;
    outB[((size_t)(b * S_ + srow)) * D_ + h * DK_ + n0 + n] = (bf16_t)acc[r];
  }
}

// ---------------------------------------------------------------------------
// Host orchestration
// ---------------------------------------------------------------------------
extern "C" void kernel_launch(void* const* d_in, const int* in_sizes, int n_in,
                              void* d_out, int out_size, void* d_ws, size_t ws_size,
                              hipStream_t stream) {
  (void)in_sizes; (void)n_in; (void)out_size; (void)ws_size;
  const float* q_embed  = (const float*)d_in[0];
  const float* qa_embed = (const float*)d_in[1];
  const float* ssim     = (const float*)d_in[2];
  const float* dec      = (const float*)d_in[3];
  const float* P1[14];
  const float* P2[14];
  for (int i = 0; i < 14; ++i) { P1[i] = (const float*)d_in[4 + i]; P2[i] = (const float*)d_in[18 + i]; }

  // workspace bump allocator
  char* wsp = (char*)d_ws;
  auto alloc = [&](size_t bytes) {
    char* p = wsp;
    wsp += (bytes + 255) & ~(size_t)255;
    return (void*)p;
  };
  const size_t actF = (size_t)MTOT * D_ * sizeof(float);
  const size_t actB = (size_t)MTOT * D_ * sizeof(bf16_t);

  float*  xF    = (float*)d_out;                 // final result chain lives here
  float*  yF    = (float*)alloc(actF);
  bf16_t* xB    = (bf16_t*)alloc(actB);
  bf16_t* yB    = (bf16_t*)alloc(actB);
  bf16_t* qpB   = (bf16_t*)alloc(actB);
  bf16_t* kB    = (bf16_t*)alloc(actB);
  bf16_t* vB    = (bf16_t*)alloc(actB);
  bf16_t* vtB   = (bf16_t*)alloc(actB);
  bf16_t* attnB = (bf16_t*)alloc(actB);
  bf16_t* ffnB  = (bf16_t*)alloc((size_t)MTOT * DF_ * sizeof(bf16_t));
  float*  tmpF  = (float*)alloc(actF);
  bf16_t* wtB   = (bf16_t*)alloc((size_t)DF_ * D_ * sizeof(bf16_t));

  auto gemm = [&](const bf16_t* A, const float* W, const float* bias,
                  int K, int N, float* outF, bf16_t* outB, int relu) {
    const size_t wn = (size_t)K * N;
    wcast_kernel<<<(unsigned)((wn + 255) / 256), 256, 0, stream>>>(W, wtB, K, N);
    dim3 grid(N / TN_, MTOT / TM_);
    gemm_bf16_kernel<<<grid, 256, 0, stream>>>(A, wtB, bias, outF, outB, MTOT, N, K, relu);
  };

  auto run_block = [&](const float* const* P, int L, float* qF, bf16_t* qB,
                       const bf16_t* kinB, const bf16_t* valB,
                       int strict, int zp, int apply_pos) {
    const float* Wk  = P[0]  + (size_t)L * D_ * D_;
    const float* bk  = P[1]  + (size_t)L * D_;
    const float* Wv  = P[2]  + (size_t)L * D_ * D_;
    const float* bv  = P[3]  + (size_t)L * D_;
    const float* Wo  = P[4]  + (size_t)L * D_ * D_;
    const float* bo  = P[5]  + (size_t)L * D_;
    const float* g1  = P[6]  + (size_t)L * D_;
    const float* be1 = P[7]  + (size_t)L * D_;
    const float* W1  = P[8]  + (size_t)L * D_ * DF_;
    const float* b1  = P[9]  + (size_t)L * DF_;
    const float* W2  = P[10] + (size_t)L * DF_ * D_;
    const float* b2  = P[11] + (size_t)L * D_;
    const float* g2  = P[12] + (size_t)L * D_;
    const float* be2 = P[13] + (size_t)L * D_;

    gemm(kinB, Wk, bk, D_, D_, nullptr, kB, 0);            // K projection
    gemm(qB,   Wk, bk, D_, D_, nullptr, qpB, 0);           // Q projection
    gemm(valB, Wv, bv, D_, D_, nullptr, vB, 0);            // V projection
    vtrans_kernel<<<(B_ * S_ * D_ + 255) / 256, 256, 0, stream>>>(vB, vtB);
    attn_kernel<<<B_ * H_ * (S_ / 16), 128, 0, stream>>>(qpB, kB, vtB, ssim, dec,
                                                         attnB, strict, zp);
    gemm(attnB, Wo, bo, D_, D_, tmpF, nullptr, 0);         // output projection
    ln_residual_kernel<<<MTOT, 128, 0, stream>>>(tmpF, qF, g1, be1, qF, qB);
    if (apply_pos) {
      gemm(qB,   W1, b1, D_, DF_, nullptr, ffnB, 1);       // FFN up + ReLU
      gemm(ffnB, W2, b2, DF_, D_, tmpF, nullptr, 0);       // FFN down
      ln_residual_kernel<<<MTOT, 128, 0, stream>>>(tmpF, qF, g2, be2, qF, qB);
    }
  };

  const size_t nact = (size_t)MTOT * D_;
  castcopy_kernel<<<(unsigned)((nact + 255) / 256), 256, 0, stream>>>(qa_embed, yF, yB, nact);
  castcopy_kernel<<<(unsigned)((nact + 255) / 256), 256, 0, stream>>>(q_embed,  xF, xB, nact);

  for (int i = 0; i < 4; ++i)
    run_block(P1, i, yF, yB, yB, yB, /*strict*/0, /*zp*/0, /*pos*/1);

  for (int i = 0; i < 8; ++i) {
    if ((i & 1) == 0)
      run_block(P2, i, xF, xB, xB, xB, 0, 0, 0);
    else
      run_block(P2, i, xF, xB, xB, yB, 1, 1, 1);
  }
}
